// SparseLinearAttentionImpl_35442070126841
// MI455X (gfx1250) — matrix operand order repaired and checked by reading, not compiled, not run on hardware
//
#include <hip/hip_runtime.h>

#define B_    2
#define L_    4096
#define H_    16
#define D_    64
#define M_    32      // L / BLKQ
#define KB_   64      // L / BLKK
#define TOPK_ 6
#define BLKQ_ 128
#define BLKK_ 64
#define EPS_  1e-5f
#define SCALE_ 0.125f // D^-0.5

typedef __attribute__((ext_vector_type(16))) _Float16 v16h;
typedef __attribute__((ext_vector_type(8)))  _Float16 v8h;
typedef __attribute__((ext_vector_type(8)))  float    v8f;

__device__ __forceinline__ int in_idx(int b, int l, int h, int d) {
    return ((b * L_ + l) * H_ + h) * D_ + d;
}

// A-operand loader: 16x32 f16 tile from row-major LDS (row stride baked into rowptr's array).
// Contiguous K order (matches B loads): lanes 0-15 hold K=0..15, lanes 16-31 hold K=16..31.
__device__ __forceinline__ v16h load_a(const _Float16* rowptr, int k0, int lane) {
    union { v16h v; v8h h[2]; } u;
    int hs = ((lane >> 4) & 1) * 16;
    u.h[0] = *(const v8h*)(rowptr + k0 + hs);
    u.h[1] = *(const v8h*)(rowptr + k0 + 8 + hs);
    return u.v;
}

__device__ __forceinline__ v8f wmma16(v16h a, v16h b, v8f c) {
    return __builtin_amdgcn_wmma_f32_16x16x32_f16(false, a, false, b, (short)0, c, false, false);
}

// ---------------------------------------------------------------------------
// Kernel 1: block-mean pooling. pq = blockmean(q, 128); pk = blockmean(k,64) - mean_L(k)
// ---------------------------------------------------------------------------
__global__ __launch_bounds__(256) void pool_kernel(const float* __restrict__ q,
                                                   const float* __restrict__ k,
                                                   float* __restrict__ pq,
                                                   float* __restrict__ pk) {
    int bh = blockIdx.x; int b = bh / H_; int h = bh % H_;
    int tid = threadIdx.x;
    __shared__ float spk[KB_][D_];
    __shared__ float smean[D_];

    for (int o = tid; o < M_ * D_; o += 256) {
        int mb = o >> 6, d = o & 63;
        float s = 0.f;
        for (int i = 0; i < BLKQ_; ++i) s += q[in_idx(b, mb * BLKQ_ + i, h, d)];
        pq[bh * M_ * D_ + o] = s * (1.f / BLKQ_);
    }
    for (int o = tid; o < KB_ * D_; o += 256) {
        int kb = o >> 6, d = o & 63;
        float s = 0.f;
        for (int i = 0; i < BLKK_; ++i) s += k[in_idx(b, kb * BLKK_ + i, h, d)];
        spk[kb][d] = s * (1.f / BLKK_);
    }
    __syncthreads();
    if (tid < D_) {
        float s = 0.f;
        for (int kb = 0; kb < KB_; ++kb) s += spk[kb][tid];
        smean[tid] = s * (1.f / KB_);
    }
    __syncthreads();
    for (int o = tid; o < KB_ * D_; o += 256)
        pk[bh * KB_ * D_ + o] = spk[o >> 6][o & 63] - smean[o & 63];
}

// ---------------------------------------------------------------------------
// Kernel 2: score row + top-6 LUT per (b,h,m). 64 threads: one key-block each.
// ---------------------------------------------------------------------------
__global__ __launch_bounds__(64) void topk_kernel(const float* __restrict__ pq,
                                                  const float* __restrict__ pk,
                                                  int* __restrict__ lut) {
    int gid = blockIdx.x;            // bh*M + m
    int m = gid & (M_ - 1); int bh = gid >> 5;
    int tid = threadIdx.x;
    __shared__ float spq[D_];
    __shared__ float sc[KB_];
    spq[tid] = pq[bh * M_ * D_ + m * D_ + tid];
    __syncthreads();
    float s = 0.f;
    const float* pkr = pk + bh * KB_ * D_ + tid * D_;
    for (int d = 0; d < D_; ++d) s += spq[d] * pkr[d];
    sc[tid] = s;
    __syncthreads();
    if (tid == 0) {
        for (int t = 0; t < TOPK_; ++t) {
            float best = -3.4e38f; int bi = 0;
            for (int kb = 0; kb < KB_; ++kb)
                if (sc[kb] > best) { best = sc[kb]; bi = kb; }
            lut[gid * TOPK_ + t] = bi;
            sc[bi] = -3.4e38f;
        }
    }
}

// ---------------------------------------------------------------------------
// Kernel 3: kv[b,h] = softmax(k)^T @ v  (64x64, K=L via WMMA), kfs = sum_l kf
// ---------------------------------------------------------------------------
#define KSTR 40   // 80B rows: 16B aligned, non power-of-two banks
__global__ __launch_bounds__(256) void kv_kernel(const float* __restrict__ k,
                                                 const float* __restrict__ v,
                                                 float* __restrict__ kv,
                                                 float* __restrict__ kfs) {
    int bh = blockIdx.x; int b = bh / H_; int h = bh % H_;
    int tid = threadIdx.x, lane = tid & 31, wave = tid >> 5;
    __shared__ __attribute__((aligned(32))) _Float16 skf[D_][KSTR]; // [d][l-in-chunk]
    __shared__ __attribute__((aligned(32))) _Float16 svt[D_][KSTR]; // [e][l-in-chunk]
    float kacc = 0.f;
    int t0 = wave * 2, t1 = wave * 2 + 1;
    int tm0 = t0 >> 2, tn0 = t0 & 3, tm1 = t1 >> 2, tn1 = t1 & 3;
    int nl = lane & 15, kcol = (lane >> 4) << 4;
    v8f c0 = {}, c1 = {};

    for (int ch = 0; ch < L_ / 32; ++ch) {
        __syncthreads();
        if (tid < 32) {                       // softmax k row -> skf[d][l] (transposed)
            int l = ch * 32 + tid;
            const float* kr = k + in_idx(b, l, h, 0);
            float mx = -3.4e38f;
            #pragma unroll
            for (int d = 0; d < D_; ++d) mx = fmaxf(mx, kr[d]);
            float e[D_]; float sum = 0.f;
            #pragma unroll
            for (int d = 0; d < D_; ++d) { e[d] = __expf(kr[d] - mx); sum += e[d]; }
            float inv = 1.f / sum;
            #pragma unroll
            for (int d = 0; d < D_; ++d) skf[d][tid] = (_Float16)(e[d] * inv);
        } else if (tid < 64) {                // v row -> svt[e][l] (transposed)
            int t = tid - 32; int l = ch * 32 + t;
            const float* vr = v + in_idx(b, l, h, 0);
            #pragma unroll
            for (int d = 0; d < D_; ++d) svt[d][t] = (_Float16)vr[d];
        }
        __syncthreads();
        if (tid < D_) {                       // kf column sums
            #pragma unroll
            for (int j = 0; j < 32; ++j) kacc += (float)skf[tid][j];
        }
        v16h a0 = load_a(&skf[tm0 * 16 + nl][0], 0, lane);
        v16h b0 = *(const v16h*)&svt[tn0 * 16 + nl][kcol];
        c0 = wmma16(a0, b0, c0);
        v16h a1 = load_a(&skf[tm1 * 16 + nl][0], 0, lane);
        v16h b1 = *(const v16h*)&svt[tn1 * 16 + nl][kcol];
        c1 = wmma16(a1, b1, c1);
    }
    int half8 = ((lane >> 4) & 1) * 8;
    #pragma unroll
    for (int j = 0; j < 8; ++j) {
        kv[bh * D_ * D_ + (tm0 * 16 + j + half8) * D_ + tn0 * 16 + nl] = c0[j];
        kv[bh * D_ * D_ + (tm1 * 16 + j + half8) * D_ + tn1 * 16 + nl] = c1[j];
    }
    if (tid < D_) kfs[bh * D_ + tid] = kacc;
}

// ---------------------------------------------------------------------------
// Kernel 4: fused block-sparse flash attention + linear branch + projection
// grid (M, H, B), 256 threads (8 waves), each wave = 16 query rows.
// ---------------------------------------------------------------------------
#define QS 72     // row stride (144B = 9*16B)
__global__ __launch_bounds__(256) void attn_kernel(const float* __restrict__ q,
                                                   const float* __restrict__ k,
                                                   const float* __restrict__ v,
                                                   const float* __restrict__ W,
                                                   const float* __restrict__ bias,
                                                   const int* __restrict__ lut,
                                                   const float* __restrict__ kv,
                                                   const float* __restrict__ kfs,
                                                   float* __restrict__ out) {
    int m = blockIdx.x, h = blockIdx.y, b = blockIdx.z;
    int bh = b * H_ + h;
    int tid = threadIdx.x, lane = tid & 31, wave = tid >> 5;

    // smem0: phase1 = sQ[128][QS]; loop = sK[64][QS] + sVT[64][QS]; phase3 = sQF[128][QS]
    __shared__ __attribute__((aligned(32))) _Float16 smem0[BLKQ_ * QS];
    __shared__ __attribute__((aligned(32))) _Float16 smemP[8 * 16 * QS]; // per-wave P/O staging
    __shared__ __attribute__((aligned(32))) _Float16 smW[D_ * QS];      // W[dout][e]
    __shared__ __attribute__((aligned(32))) _Float16 smKVT[D_ * QS];    // kv^T [e][d]
    __shared__ float sDen[BLKQ_];
    __shared__ float sKFS[D_];
    __shared__ float sBias[D_];

    if (tid < D_) { sKFS[tid] = kfs[bh * D_ + tid]; sBias[tid] = bias[tid]; }
    if (tid < BLKQ_) {                       // raw Q -> f16
        const float* qr = q + in_idx(b, m * BLKQ_ + tid, h, 0);
        for (int d = 0; d < D_; ++d) smem0[tid * QS + d] = (_Float16)qr[d];
    } else {
        int t = tid - 128;
        for (int i = t; i < D_ * D_; i += 128) smW[(i >> 6) * QS + (i & 63)] = (_Float16)W[i];
        for (int i = t; i < D_ * D_; i += 128) smKVT[(i & 63) * QS + (i >> 6)] = (_Float16)kv[bh * D_ * D_ + i];
    }
    __syncthreads();

    int r0 = wave * 16;
    int nl = lane & 15, kcol = (lane >> 4) << 4, half8 = ((lane >> 4) & 1) * 8;
    v16h aq0 = load_a(&smem0[(r0 + nl) * QS], 0, lane);   // Q resident in regs
    v16h aq1 = load_a(&smem0[(r0 + nl) * QS], 32, lane);
    __syncthreads();                                       // sQ dead -> reuse as K/VT

    float mrun[8], lrun[8];
    v8f O[4] = {{}, {}, {}, {}};
    #pragma unroll
    for (int j = 0; j < 8; ++j) { mrun[j] = -3.4e38f; lrun[j] = 0.f; }

    _Float16* sK  = smem0;               // [64][QS] key-major
    _Float16* sVT = smem0 + 64 * QS;     // [64][QS] d-major (transposed V)
    _Float16* sP  = smemP + wave * 16 * QS;
    const int* lrow = lut + (bh * M_ + m) * TOPK_;

    for (int it = 0; it < TOPK_; ++it) {
        int kb = lrow[it];
        if (it + 1 < TOPK_) {            // prefetch next selected block into L2
            int kbn = lrow[it + 1];
            __builtin_prefetch(k + in_idx(b, kbn * BLKK_ + (tid >> 2), h, (tid & 3) * 16), 0, 1);
            __builtin_prefetch(v + in_idx(b, kbn * BLKK_ + (tid >> 2), h, (tid & 3) * 16), 0, 1);
        }
        __syncthreads();
        for (int i = tid; i < BLKK_ * D_; i += 256) {
            int key = i >> 6, d = i & 63;
            float kvv = k[in_idx(b, kb * BLKK_ + key, h, d)];
            float vvv = v[in_idx(b, kb * BLKK_ + key, h, d)];
            sK [key * QS + d]  = (_Float16)kvv;
            sVT[d * QS + key]  = (_Float16)vvv;
        }
        __syncthreads();

        // S = Q K^T : 4 tiles of 16 keys, K-dim = d (2 x k32)
        v8f S[4];
        #pragma unroll
        for (int t = 0; t < 4; ++t) {
            v8f s = {};
            s = wmma16(aq0, *(const v16h*)&sK[(t * 16 + nl) * QS + kcol], s);
            s = wmma16(aq1, *(const v16h*)&sK[(t * 16 + nl) * QS + 32 + kcol], s);
            S[t] = s * SCALE_;
        }
        // online softmax (rows spread: VGPR j = row j/half8, lanes = cols)
        #pragma unroll
        for (int j = 0; j < 8; ++j) {
            float x = fmaxf(fmaxf(S[0][j], S[1][j]), fmaxf(S[2][j], S[3][j]));
            #pragma unroll
            for (int o = 1; o < 16; o <<= 1) x = fmaxf(x, __shfl_xor(x, o, 32));
            float mn = fmaxf(mrun[j], x);
            float corr = __expf(mrun[j] - mn);
            float s = 0.f;
            #pragma unroll
            for (int t = 0; t < 4; ++t) { float p = __expf(S[t][j] - mn); S[t][j] = p; s += p; }
            #pragma unroll
            for (int o = 1; o < 16; o <<= 1) s += __shfl_xor(s, o, 32);
            lrun[j] = lrun[j] * corr + s;
            mrun[j] = mn;
            #pragma unroll
            for (int t = 0; t < 4; ++t) O[t][j] *= corr;
        }
        // stage P -> wave-private LDS, reload in A layout
        #pragma unroll
        for (int j = 0; j < 8; ++j)
            #pragma unroll
            for (int t = 0; t < 4; ++t)
                sP[(j + half8) * QS + t * 16 + nl] = (_Float16)S[t][j];
        asm volatile("s_wait_dscnt 0" ::: "memory");
        v16h ap0 = load_a(&sP[nl * QS], 0, lane);
        v16h ap1 = load_a(&sP[nl * QS], 32, lane);
        #pragma unroll
        for (int t = 0; t < 4; ++t) {
            O[t] = wmma16(ap0, *(const v16h*)&sVT[(t * 16 + nl) * QS + kcol], O[t]);
            O[t] = wmma16(ap1, *(const v16h*)&sVT[(t * 16 + nl) * QS + 32 + kcol], O[t]);
        }
    }
    #pragma unroll
    for (int j = 0; j < 8; ++j) {
        float invl = 1.f / lrun[j];
        #pragma unroll
        for (int t = 0; t < 4; ++t) O[t][j] *= invl;
    }

    // ---- linear branch: qf staged into (now free) smem0 ----
    __syncthreads();
    if (tid < BLKQ_) {
        const float* qr = q + in_idx(b, m * BLKQ_ + tid, h, 0);
        float mx = -3.4e38f;
        for (int d = 0; d < D_; ++d) mx = fmaxf(mx, qr[d]);
        float sum = 0.f;
        for (int d = 0; d < D_; ++d) { float e = __expf(qr[d] - mx); smem0[tid * QS + d] = (_Float16)e; sum += e; }
        float inv = 1.f / sum, den = 0.f;
        for (int d = 0; d < D_; ++d) {
            float p = (float)smem0[tid * QS + d] * inv;
            smem0[tid * QS + d] = (_Float16)p;
            den += p * sKFS[d];
        }
        sDen[tid] = den;
    }
    __syncthreads();

    v16h af0 = load_a(&smem0[(r0 + nl) * QS], 0, lane);
    v16h af1 = load_a(&smem0[(r0 + nl) * QS], 32, lane);
    v8f NUM[4];
    #pragma unroll
    for (int t = 0; t < 4; ++t) {
        v8f s = {};
        s = wmma16(af0, *(const v16h*)&smKVT[(t * 16 + nl) * QS + kcol], s);
        s = wmma16(af1, *(const v16h*)&smKVT[(t * 16 + nl) * QS + 32 + kcol], s);
        NUM[t] = s;
    }
    #pragma unroll
    for (int j = 0; j < 8; ++j) {
        float dj = 1.f / (EPS_ + sDen[r0 + j + half8]);
        #pragma unroll
        for (int t = 0; t < 4; ++t)
            sP[(j + half8) * QS + t * 16 + nl] = (_Float16)(NUM[t][j] * dj);
    }
    asm volatile("s_wait_dscnt 0" ::: "memory");
    v16h ao0 = load_a(&sP[nl * QS], 0, lane);
    v16h ao1 = load_a(&sP[nl * QS], 32, lane);
    #pragma unroll
    for (int t = 0; t < 4; ++t) {
        v8f s = {};
        s = wmma16(ao0, *(const v16h*)&smW[(t * 16 + nl) * QS + kcol], s);
        s = wmma16(ao1, *(const v16h*)&smW[(t * 16 + nl) * QS + 32 + kcol], s);
        #pragma unroll
        for (int j = 0; j < 8; ++j) {
            int lr = m * BLKQ_ + r0 + j + half8;
            out[in_idx(b, lr, h, t * 16 + nl)] = O[t][j] + s[j] + sBias[t * 16 + nl];
        }
    }
}

// ---------------------------------------------------------------------------
extern "C" void kernel_launch(void* const* d_in, const int* in_sizes, int n_in,
                              void* d_out, int out_size, void* d_ws, size_t ws_size,
                              hipStream_t stream) {
    (void)in_sizes; (void)n_in; (void)out_size; (void)ws_size;
    const float* q  = (const float*)d_in[0];
    const float* k  = (const float*)d_in[1];
    const float* v  = (const float*)d_in[2];
    const float* W  = (const float*)d_in[3];
    const float* bv = (const float*)d_in[4];
    float* out = (float*)d_out;

    float* ws  = (float*)d_ws;
    float* pq  = ws;                       // B*H*M*D   = 65536
    float* pk  = pq + B_ * H_ * M_ * D_;   // B*H*KB*D  = 131072
    int*   lut = (int*)(pk + B_ * H_ * KB_ * D_);          // B*H*M*TOPK = 6144
    float* kv  = (float*)(lut + B_ * H_ * M_ * TOPK_);     // B*H*D*D   = 131072
    float* kfs = kv + B_ * H_ * D_ * D_;                   // B*H*D     = 2048

    pool_kernel<<<dim3(B_ * H_), dim3(256), 0, stream>>>(q, k, pq, pk);
    topk_kernel<<<dim3(B_ * H_ * M_), dim3(64), 0, stream>>>(pq, pk, lut);
    kv_kernel  <<<dim3(B_ * H_), dim3(256), 0, stream>>>(k, v, kv, kfs);
    attn_kernel<<<dim3(M_, H_, B_), dim3(256), 0, stream>>>(q, k, v, W, bv, lut, kv, kfs, out);
}
